// BudgetedDetailTokenizer_37048387895854
// MI455X (gfx1250) — compile-verified, matrix operand order checked by reading, not run on hardware
//
#include <hip/hip_runtime.h>
#include <math.h>

// ---------------- CDNA5 WMMA types / helpers ----------------
typedef __attribute__((ext_vector_type(16))) __bf16 v16bf;
typedef __attribute__((ext_vector_type(8)))  __bf16 v8bf;
typedef __attribute__((ext_vector_type(8)))  float  v8f;

__device__ __forceinline__ v8f wmma_bf16(v16bf a, v16bf b, v8f c) {
    // D = A(16x32 bf16) * B(32x16 bf16) + C(16x16 f32)
    return __builtin_amdgcn_wmma_f32_16x16x32_bf16(
        false, a, false, b, (short)0, c, false, false);
}

__device__ __forceinline__ v8f vzero8() {
    v8f z;
#pragma unroll
    for (int i = 0; i < 8; ++i) z[i] = 0.0f;
    return z;
}

// Two contiguous 16-byte chunks -> one 16-element bf16 operand register set.
__device__ __forceinline__ v16bf load2x8(const __bf16* __restrict__ p0,
                                         const __bf16* __restrict__ p1) {
    v8bf lo = *(const v8bf*)p0;
    v8bf hi = *(const v8bf*)p1;
    v16bf r;
#pragma unroll
    for (int i = 0; i < 8; ++i) { r[i] = lo[i]; r[i + 8] = hi[i]; }
    return r;
}

__device__ __forceinline__ __bf16 f2bf(float f) {
    union { float f; unsigned u; } v; v.f = f;
    unsigned r = (v.u + 0x7FFFu + ((v.u >> 16) & 1u)) >> 16;
    unsigned short s = (unsigned short)r;
    __bf16 o; __builtin_memcpy(&o, &s, 2);
    return o;
}
__device__ __forceinline__ float bf2f(__bf16 b) {
    unsigned short s; __builtin_memcpy(&s, &b, 2);
    union { unsigned u; float f; } v; v.u = ((unsigned)s) << 16;
    return v.f;
}
__device__ __forceinline__ float gelu(float x) {
    return 0.5f * x * (1.0f + erff(x * 0.70710678118654752f));
}

// ---------------- problem constants ----------------
#define BATCH 16
#define H 384
#define HP 386          // padded
#define C 64
#define E 768
#define G 24            // coarse grid
#define NT 576          // G*G tokens per image
static const float BN_EPS = 1e-5f;

// ---------------- workspace layout (bytes, all 16B aligned) ----------------
#define N_F1 ((size_t)BATCH * HP * HP * C)            // bf16, zero-padded NHWC
#define N_F2 ((size_t)BATCH * H * H * C)              // bf16 NHWC
#define N_DM ((size_t)BATCH * NT * 4 * E)             // bf16, dp layout (b,t,(dy,dx,e))
#define O_F1 ((size_t)0)
#define O_F2 (O_F1 + N_F1 * 2)
#define O_DM (O_F2 + N_F2 * 2)
#define O_W2 (O_DM + N_DM * 2)                        // [64][576] bf16
#define O_WC (O_W2 + (size_t)64 * 576 * 2)            // [768][16384] bf16
#define O_WD (O_WC + (size_t)768 * 16384 * 2)         // [768][4096] bf16
#define O_WM (O_WD + (size_t)768 * 4096 * 2)          // [768][3072] bf16
#define O_SC (O_WM + (size_t)768 * 3072 * 2)          // [B][576] f32
#define O_IX (O_SC + (size_t)BATCH * NT * 4)          // [B][64] i32

#define COARSE_N ((size_t)BATCH * NT * E)
#define DETAIL_N ((size_t)BATCH * 64 * E)

// =====================================================================
// Stem conv1: 3x3, 3->64, SAME + BN + GELU -> padded NHWC bf16 f1
// one pixel per block, 64 threads = 64 output channels
// =====================================================================
__global__ __launch_bounds__(64) void k_conv1(
    const float* __restrict__ x, const float* __restrict__ w,
    const float* __restrict__ g, const float* __restrict__ bb,
    const float* __restrict__ m, const float* __restrict__ vv,
    __bf16* __restrict__ f1) {
    __shared__ float patch[27];
    __shared__ float wsm[64 * 27];
    int pix = blockIdx.x;
    int b = pix / (H * H);
    int r = pix % (H * H);
    int y = r / H, xx = r % H;
    int t = threadIdx.x;
#pragma unroll
    for (int i = t; i < 64 * 27; i += 64) wsm[i] = w[i]; // OIHW contiguous per co
    if (t < 27) {
        int ci = t / 9, ky = (t % 9) / 3, kx = t % 3;
        int yy = y + ky - 1, xw = xx + kx - 1;
        float val = 0.0f;
        if (yy >= 0 && yy < H && xw >= 0 && xw < H)
            val = x[((size_t)(b * 3 + ci) * H + yy) * H + xw];
        patch[t] = val;
    }
    __syncthreads();
    float s = 0.0f;
#pragma unroll
    for (int i = 0; i < 27; ++i) s += patch[i] * wsm[t * 27 + i];
    float sc = g[t] * rsqrtf(vv[t] + BN_EPS);
    float val = gelu((s - m[t]) * sc + bb[t]);
    f1[((size_t)(b * HP + y + 1) * HP + (xx + 1)) * C + t] = f2bf(val);
}

// =====================================================================
// Weight reorder+convert kernels (f32 OIHW -> [E][K] bf16, K = (py,px,c))
// =====================================================================
__global__ void k_w2r(const float* __restrict__ w, __bf16* __restrict__ o) {
    int i = blockIdx.x * blockDim.x + threadIdx.x;
    if (i >= 64 * 576) return;
    int e = i / 576, k = i % 576;
    int p = k >> 6, c = k & 63;          // p = ky*3+kx
    int ky = p / 3, kx = p % 3;
    o[i] = f2bf(w[(((size_t)e * 64 + c) * 3 + ky) * 3 + kx]);
}
__global__ void k_wcr(const float* __restrict__ w, __bf16* __restrict__ o) {
    size_t i = (size_t)blockIdx.x * blockDim.x + threadIdx.x;
    if (i >= (size_t)768 * 16384) return;
    int e = (int)(i >> 14), k = (int)(i & 16383);
    int p = k >> 6, c = k & 63;          // p = py*16+px
    int py = p >> 4, px = p & 15;
    o[i] = f2bf(w[(((size_t)e * 64 + c) * 16 + py) * 16 + px]);
}
__global__ void k_wdr(const float* __restrict__ w, __bf16* __restrict__ o) {
    size_t i = (size_t)blockIdx.x * blockDim.x + threadIdx.x;
    if (i >= (size_t)768 * 4096) return;
    int e = (int)(i >> 12), k = (int)(i & 4095);
    int p = k >> 6, c = k & 63;          // p = py*8+px
    int py = p >> 3, px = p & 7;
    o[i] = f2bf(w[(((size_t)e * 64 + c) * 8 + py) * 8 + px]);
}
__global__ void k_wmr(const float* __restrict__ w, __bf16* __restrict__ o) {
    size_t i = (size_t)blockIdx.x * blockDim.x + threadIdx.x;
    if (i >= (size_t)768 * 3072) return;
    o[i] = f2bf(w[i]);
}

// =====================================================================
// conv2: implicit GEMM via WMMA. M = B*384*384 pixels, N = 64, K = 576.
// block = 4 waves; wave w owns 16 output channels. + BN + GELU -> f2 NHWC bf16
// =====================================================================
__global__ __launch_bounds__(128) void k_conv2(
    const __bf16* __restrict__ f1, const __bf16* __restrict__ w2r,
    const float* __restrict__ g, const float* __restrict__ bb,
    const float* __restrict__ m, const float* __restrict__ vv,
    __bf16* __restrict__ f2) {
    int mt = blockIdx.x;                       // B*384*24 tiles of 16 px
    int b = mt / (H * (H / 16));
    int r = mt % (H * (H / 16));
    int y = r / (H / 16);
    int xt = (r % (H / 16)) * 16;
    int lane = threadIdx.x & 31, wave = threadIdx.x >> 5;
    int ln = lane & 15, hf = lane >> 4;
    int eBase = wave * 16;

    v8f c = vzero8();
    for (int k0 = 0; k0 < 576; k0 += 32) {
        int kk = k0 >> 6;                      // (ky,kx) fixed across this K-slice
        int ky = kk / 3, kx = kk % 3;
        int cb = (k0 & 63) + hf * 8;
        const __bf16* pA = f1 +
            ((size_t)(b * HP + y + ky) * HP + (xt + ln + kx)) * C + cb;
        v16bf a = load2x8(pA, pA + 16);
        const __bf16* pB = w2r + (size_t)(eBase + ln) * 576 + k0 + hf * 8;
        v16bf bm = load2x8(pB, pB + 16);
        c = wmma_bf16(a, bm, c);
    }
    int e = eBase + ln;
    float sc = g[e] * rsqrtf(vv[e] + BN_EPS);
    float sh = bb[e] - m[e] * sc;
#pragma unroll
    for (int rr = 0; rr < 8; ++rr) {
        int M = rr + hf * 8;
        float val = gelu(c[rr] * sc + sh);
        f2[((size_t)(b * H + y) * H + (xt + M)) * C + e] = f2bf(val);
    }
}

// =====================================================================
// coarse: GEMM [B*576 x 768 x 16384]; wave owns 4 N-tiles (A reuse x4)
// writes coarse_tokens (f32) straight into d_out region 0
// =====================================================================
__global__ __launch_bounds__(128) void k_coarse(
    const __bf16* __restrict__ f2, const __bf16* __restrict__ wc,
    const float* __restrict__ bias, float* __restrict__ out) {
    int mt = blockIdx.x;                       // 576 M-tiles
    int lane = threadIdx.x & 31, wave = threadIdx.x >> 5;
    int ln = lane & 15, hf = lane >> 4;
    int T = mt * 16 + ln;
    int b = T / NT, t = T % NT, gy = t / G, gx = t % G;
    int eg = (blockIdx.y * 4 + wave) * 64;     // 64 channels per wave

    v8f c[4];
#pragma unroll
    for (int n = 0; n < 4; ++n) c[n] = vzero8();

    for (int k0 = 0; k0 < 16384; k0 += 32) {
        int p = k0 >> 6;                       // py*16+px
        int py = p >> 4, px = p & 15;
        int cb = (k0 & 63) + hf * 8;
        const __bf16* pA = f2 +
            ((size_t)(b * H + gy * 16 + py) * H + (gx * 16 + px)) * C + cb;
        __builtin_prefetch(pA + (size_t)H * C, 0, 1);  // next K-slice row
        v16bf a = load2x8(pA, pA + 16);
#pragma unroll
        for (int n = 0; n < 4; ++n) {
            const __bf16* pB = wc + (size_t)(eg + n * 16 + ln) * 16384 + k0 + hf * 8;
            v16bf bm = load2x8(pB, pB + 16);
            c[n] = wmma_bf16(a, bm, c[n]);
        }
    }
#pragma unroll
    for (int n = 0; n < 4; ++n) {
        int e = eg + n * 16 + ln;
        float bs = bias[e];
#pragma unroll
        for (int rr = 0; rr < 8; ++rr) {
            int TM = mt * 16 + rr + hf * 8;
            out[(size_t)TM * E + e] = c[n][rr] + bs;
        }
    }
}

// =====================================================================
// detail: GEMM [B*48*48 x 768 x 4096] -> dm bf16 in dp layout (b,t,(dy,dx,e))
// =====================================================================
__global__ __launch_bounds__(128) void k_detail(
    const __bf16* __restrict__ f2, const __bf16* __restrict__ wd,
    const float* __restrict__ bias, __bf16* __restrict__ dm) {
    int mt = blockIdx.x;                       // B*48*3 tiles
    int b = mt / (48 * 3);
    int r = mt % (48 * 3);
    int Y = r / 3;
    int xt = (r % 3) * 16;
    int lane = threadIdx.x & 31, wave = threadIdx.x >> 5;
    int ln = lane & 15, hf = lane >> 4;
    int X = xt + ln;
    int eg = (blockIdx.y * 4 + wave) * 64;

    v8f c[4];
#pragma unroll
    for (int n = 0; n < 4; ++n) c[n] = vzero8();

    for (int k0 = 0; k0 < 4096; k0 += 32) {
        int p = k0 >> 6;                       // py*8+px
        int py = p >> 3, px = p & 7;
        int cb = (k0 & 63) + hf * 8;
        const __bf16* pA = f2 +
            ((size_t)(b * H + Y * 8 + py) * H + (X * 8 + px)) * C + cb;
        v16bf a = load2x8(pA, pA + 16);
#pragma unroll
        for (int n = 0; n < 4; ++n) {
            const __bf16* pB = wd + (size_t)(eg + n * 16 + ln) * 4096 + k0 + hf * 8;
            v16bf bm = load2x8(pB, pB + 16);
            c[n] = wmma_bf16(a, bm, c[n]);
        }
    }
    int gy = Y >> 1, dy = Y & 1;
#pragma unroll
    for (int n = 0; n < 4; ++n) {
        int e = eg + n * 16 + ln;
        float bs = bias[e];
#pragma unroll
        for (int rr = 0; rr < 8; ++rr) {
            int Xs = xt + rr + hf * 8;
            int gx = Xs >> 1, dx = Xs & 1;
            size_t off = (((size_t)(b * G + gy) * G + gx) * 4 + dy * 2 + dx) * E + e;
            dm[off] = f2bf(c[n][rr] + bs);
        }
    }
}

// =====================================================================
// variance score per 16x16 coarse cell (mean over C of per-channel var)
// =====================================================================
__global__ __launch_bounds__(256) void k_var(const __bf16* __restrict__ f2,
                                             float* __restrict__ scores) {
    __shared__ float ss[256], ss2[256];
    int bt = blockIdx.x;                       // b*576 + t
    int b = bt / NT, t = bt % NT;
    int gy = t / G, gx = t % G;
    int tid = threadIdx.x;
    int ch = tid & 63, q = tid >> 6;
    float s = 0.0f, s2 = 0.0f;
#pragma unroll 4
    for (int i = 0; i < 64; ++i) {
        int p = q * 64 + i;
        int py = p >> 4, px = p & 15;
        float v = bf2f(f2[((size_t)(b * H + gy * 16 + py) * H + gx * 16 + px) * C + ch]);
        s += v; s2 += v * v;
    }
    ss[tid] = s; ss2[tid] = s2;
    __syncthreads();
    if (tid < 64) {
        float a  = ss[tid]  + ss[tid + 64]  + ss[tid + 128]  + ss[tid + 192];
        float a2 = ss2[tid] + ss2[tid + 64] + ss2[tid + 128] + ss2[tid + 192];
        float mean = a * (1.0f / 256.0f);
        ss[tid] = a2 * (1.0f / 256.0f) - mean * mean;
    }
    __syncthreads();
    for (int w = 32; w > 0; w >>= 1) {
        if (tid < w) ss[tid] += ss[tid + w];
        __syncthreads();
    }
    if (tid == 0) scores[bt] = ss[0] * (1.0f / 64.0f);
}

// =====================================================================
// top-64 of 576 per batch (descending, lowest-index tie-break like lax.top_k)
// =====================================================================
__global__ __launch_bounds__(64) void k_topk(const float* __restrict__ scores,
                                             int* __restrict__ idxout,
                                             float* __restrict__ idxf) {
    __shared__ float sc[NT];
    __shared__ float bv[64];
    __shared__ int   bi[64];
    int b = blockIdx.x, t = threadIdx.x;
    for (int i = t; i < NT; i += 64) sc[i] = scores[b * NT + i];
    __syncthreads();
    for (int k = 0; k < 64; ++k) {
        float best = -INFINITY; int bidx = NT;
        for (int i = t; i < NT; i += 64) {
            float v = sc[i];
            if (v > best || (v == best && i < bidx)) { best = v; bidx = i; }
        }
        bv[t] = best; bi[t] = bidx;
        __syncthreads();
        if (t == 0) {
            float Bv = -INFINITY; int BI = NT;
            for (int i = 0; i < 64; ++i)
                if (bv[i] > Bv || (bv[i] == Bv && bi[i] < BI)) { Bv = bv[i]; BI = bi[i]; }
            idxout[b * 64 + k] = BI;
            idxf[b * 64 + k] = (float)BI;
            sc[BI] = -INFINITY;
        }
        __syncthreads();
    }
}

// =====================================================================
// merge: GEMM on gathered rows only [B*64 x 768 x 3072] -> detail_tokens f32
// =====================================================================
__global__ __launch_bounds__(128) void k_merge(
    const __bf16* __restrict__ dm, const __bf16* __restrict__ wm,
    const float* __restrict__ bias, const int* __restrict__ idx,
    float* __restrict__ out) {
    int mt = blockIdx.x;                       // 64 M-tiles of 16 rows
    int lane = threadIdx.x & 31, wave = threadIdx.x >> 5;
    int ln = lane & 15, hf = lane >> 4;
    int R = mt * 16 + ln;
    int b = R >> 6, j = R & 63;
    int t = idx[b * 64 + j];
    const __bf16* aRow = dm + ((size_t)b * NT + t) * 3072;
    int eg = (blockIdx.y * 4 + wave) * 64;

    v8f c[4];
#pragma unroll
    for (int n = 0; n < 4; ++n) c[n] = vzero8();

    for (int k0 = 0; k0 < 3072; k0 += 32) {
        const __bf16* pA = aRow + k0 + hf * 8;
        v16bf a = load2x8(pA, pA + 16);
#pragma unroll
        for (int n = 0; n < 4; ++n) {
            const __bf16* pB = wm + (size_t)(eg + n * 16 + ln) * 3072 + k0 + hf * 8;
            v16bf bm = load2x8(pB, pB + 16);
            c[n] = wmma_bf16(a, bm, c[n]);
        }
    }
#pragma unroll
    for (int n = 0; n < 4; ++n) {
        int e = eg + n * 16 + ln;
        float bs = bias[e];
#pragma unroll
        for (int rr = 0; rr < 8; ++rr) {
            int RM = mt * 16 + rr + hf * 8;
            out[(size_t)RM * E + e] = c[n][rr] + bs;
        }
    }
}

// =====================================================================
extern "C" void kernel_launch(void* const* d_in, const int* in_sizes, int n_in,
                              void* d_out, int out_size, void* d_ws, size_t ws_size,
                              hipStream_t stream) {
    const float* x        = (const float*)d_in[0];
    const float* conv1_w  = (const float*)d_in[1];
    const float* bn1_g    = (const float*)d_in[2];
    const float* bn1_b    = (const float*)d_in[3];
    const float* bn1_m    = (const float*)d_in[4];
    const float* bn1_v    = (const float*)d_in[5];
    const float* conv2_w  = (const float*)d_in[6];
    const float* bn2_g    = (const float*)d_in[7];
    const float* bn2_b    = (const float*)d_in[8];
    const float* bn2_m    = (const float*)d_in[9];
    const float* bn2_v    = (const float*)d_in[10];
    const float* coarse_w = (const float*)d_in[11];
    const float* coarse_b = (const float*)d_in[12];
    const float* detail_w = (const float*)d_in[13];
    const float* detail_b = (const float*)d_in[14];
    const float* merge_w  = (const float*)d_in[15];
    const float* merge_b  = (const float*)d_in[16];

    char* ws = (char*)d_ws;
    __bf16* f1  = (__bf16*)(ws + O_F1);
    __bf16* f2  = (__bf16*)(ws + O_F2);
    __bf16* dm  = (__bf16*)(ws + O_DM);
    __bf16* w2r = (__bf16*)(ws + O_W2);
    __bf16* wcr = (__bf16*)(ws + O_WC);
    __bf16* wdr = (__bf16*)(ws + O_WD);
    __bf16* wmr = (__bf16*)(ws + O_WM);
    float*  scr = (float*)(ws + O_SC);
    int*    ixb = (int*)(ws + O_IX);

    float* outCoarse = (float*)d_out;
    float* outDetail = outCoarse + COARSE_N;
    float* outIdx    = outDetail + DETAIL_N;

    // zero f1 (padding border must be 0)
    hipMemsetAsync(f1, 0, N_F1 * 2, stream);

    // weight reorder/convert
    k_w2r<<<(64 * 576 + 255) / 256, 256, 0, stream>>>(conv2_w, w2r);
    k_wcr<<<(int)(((size_t)768 * 16384 + 255) / 256), 256, 0, stream>>>(coarse_w, wcr);
    k_wdr<<<(int)(((size_t)768 * 4096 + 255) / 256), 256, 0, stream>>>(detail_w, wdr);
    k_wmr<<<(int)(((size_t)768 * 3072 + 255) / 256), 256, 0, stream>>>(merge_w, wmr);

    // stem
    k_conv1<<<BATCH * H * H, 64, 0, stream>>>(x, conv1_w, bn1_g, bn1_b, bn1_m, bn1_v, f1);
    k_conv2<<<BATCH * H * (H / 16), 128, 0, stream>>>(f1, w2r, bn2_g, bn2_b, bn2_m, bn2_v, f2);

    // tokens
    k_coarse<<<dim3(BATCH * NT / 16, 3), 128, 0, stream>>>(f2, wcr, coarse_b, outCoarse);
    k_detail<<<dim3(BATCH * 48 * 3, 3), 128, 0, stream>>>(f2, wdr, detail_b, dm);

    // scores + topk
    k_var<<<BATCH * NT, 256, 0, stream>>>(f2, scr);
    k_topk<<<BATCH, 64, 0, stream>>>(scr, ixb, outIdx);

    // merge only gathered rows
    k_merge<<<dim3(BATCH * 64 / 16, 3), 128, 0, stream>>>(dm, wmr, merge_b, ixb, outDetail);
}
